// Attention_4140348473696
// MI455X (gfx1250) — compile-verified, matrix operand order checked by reading, not run on hardware
//
#include <hip/hip_runtime.h>
#include <stdint.h>

typedef __attribute__((ext_vector_type(16))) __bf16 v16bf;
typedef __attribute__((ext_vector_type(8)))  float  v8f;
typedef __attribute__((ext_vector_type(4)))  float  vf4;

union FragA { vf4 q[2]; v16bf v; };
union FragC { v8f v; float f[8]; };

#define HEADS 16
#define DH    64
#define CMD   1024
#define SEQ   2048
#define BATCH 4
#define NTOK  (BATCH*SEQ)   // 8192

// CDNA5 async global->LDS copy (ASYNCcnt), GV addressing mode.
// lds = byte offset within workgroup LDS (low 32 bits of generic pointer),
// g   = 64-bit global address.
__device__ __forceinline__ void async_b128(unsigned lds, const void* g) {
  asm volatile("global_load_async_to_lds_b128 %0, %1, off"
               :: "v"(lds), "v"(g) : "memory");
}
__device__ __forceinline__ void wait_async0() {
  asm volatile("s_wait_asynccnt 0" ::: "memory");
}
__device__ __forceinline__ unsigned lds_off(const void* p) {
  return (unsigned)(uintptr_t)p;   // LDS aperture: addr[31:0] is the LDS byte address
}

// ---------------------------------------------------------------- convert x
__global__ __launch_bounds__(256) void cvt_f32_bf16(const float* __restrict__ in,
                                                    __bf16* __restrict__ out) {
  int i = (blockIdx.x * 256 + threadIdx.x) * 8;
  vf4 a = *(const vf4*)(in + i);
  vf4 b = *(const vf4*)(in + i + 4);
  union { __bf16 h[8]; vf4 v; } u;
  u.h[0] = (__bf16)a[0]; u.h[1] = (__bf16)a[1];
  u.h[2] = (__bf16)a[2]; u.h[3] = (__bf16)a[3];
  u.h[4] = (__bf16)b[0]; u.h[5] = (__bf16)b[1];
  u.h[6] = (__bf16)b[2]; u.h[7] = (__bf16)b[3];
  *(vf4*)(out + i) = u.v;
}

// --------------------------------------------- weight transpose + convert
// Wt[n][k] = (bf16) W[k][n]   (1024x1024), 32x32 LDS tiles
__global__ __launch_bounds__(256) void transpose_cvt(const float* __restrict__ W,
                                                     __bf16* __restrict__ Wt) {
  __shared__ __bf16 tile[32][33];
  int bx = blockIdx.x & 31;      // n tile
  int by = blockIdx.x >> 5;      // k tile
  int tx = threadIdx.x & 31;
  int ty0 = (threadIdx.x >> 5) * 4;
  for (int i = 0; i < 4; i++) {
    int ty = ty0 + i;
    tile[ty][tx] = (__bf16)W[(size_t)(by*32 + ty) * CMD + bx*32 + tx];
  }
  __syncthreads();
  for (int i = 0; i < 4; i++) {
    int ty = ty0 + i;
    Wt[(size_t)(bx*32 + ty) * CMD + by*32 + tx] = tile[tx][ty];
  }
}

// ---------------------------------------------------------------- GEMM
// C[M=8192, N=1024] = A[M,K=1024] (bf16, row major) x Wt[N,K] (bf16)
// mode 0: scatter bf16 to [b,h,s,d]   (Q, K)
// mode 1: scatter bf16 to [b,h,d,s]   (V transposed)
// mode 2: fp32 row-major + bias       (final projection)
__global__ __launch_bounds__(256) void gemm_bf16(
    const __bf16* __restrict__ A, const __bf16* __restrict__ Wt,
    __bf16* __restrict__ outb, float* __restrict__ outf,
    const float* __restrict__ bias, int mode)
{
  __shared__ alignas(16) __bf16 As[2][128][40];   // +8 pad: bank-conflict break
  __shared__ alignas(16) __bf16 Bs[2][128][40];
  const int bm = blockIdx.x >> 3;              // M tile (64 of them)
  const int bn = blockIdx.x & 7;               // N tile (8 of them)
  const int t = threadIdx.x;
  const int lane = t & 31, wave = t >> 5;      // 8 waves
  const int wm = wave >> 1, wn = wave & 1;     // 4x2 wave grid
  const int half = lane >> 4, lm = lane & 15;

  FragC acc[2][4];
  for (int i = 0; i < 2; i++)
    for (int j = 0; j < 4; j++)
      for (int r = 0; r < 8; r++) acc[i][j].f[r] = 0.f;

  const int lr = t >> 2;            // 0..63
  const int lc = (t & 3) * 8;       // 0,8,16,24
  const __bf16* Arow0 = A  + (size_t)(bm*128 + lr) * CMD;
  const __bf16* Arow1 = Arow0 + (size_t)64 * CMD;
  const __bf16* Brow0 = Wt + (size_t)(bn*128 + lr) * CMD;
  const __bf16* Brow1 = Brow0 + (size_t)64 * CMD;

  const unsigned rowStride = 40 * sizeof(__bf16);        // 80 B
  unsigned ldsA[2], ldsB[2];
  ldsA[0] = lds_off(&As[0][lr][lc]);  ldsA[1] = lds_off(&As[1][lr][lc]);
  ldsB[0] = lds_off(&Bs[0][lr][lc]);  ldsB[1] = lds_off(&Bs[1][lr][lc]);

  auto issue_tile = [&](int buf, int k0) {
    async_b128(ldsA[buf],                 Arow0 + k0 + lc);
    async_b128(ldsA[buf] + 64*rowStride,  Arow1 + k0 + lc);
    async_b128(ldsB[buf],                 Brow0 + k0 + lc);
    async_b128(ldsB[buf] + 64*rowStride,  Brow1 + k0 + lc);
  };

  issue_tile(0, 0);
  wait_async0();
  __syncthreads();

  for (int k0 = 0; k0 < CMD; k0 += 32) {
    const int cur = (k0 >> 5) & 1;
    if (k0 + 32 < CMD) issue_tile(cur ^ 1, k0 + 32);   // overlap next copy

    FragA af[2], bfr[4];
    for (int mf = 0; mf < 2; mf++) {
      const __bf16* p = &As[cur][wm*32 + mf*16 + lm][half*8];
      af[mf].q[0] = *(const vf4*)p;
      af[mf].q[1] = *(const vf4*)(p + 16);
    }
    for (int nf = 0; nf < 4; nf++) {
      const __bf16* p = &Bs[cur][wn*64 + nf*16 + lm][half*8];
      bfr[nf].q[0] = *(const vf4*)p;
      bfr[nf].q[1] = *(const vf4*)(p + 16);
    }
    for (int mf = 0; mf < 2; mf++)
      for (int nf = 0; nf < 4; nf++)
        acc[mf][nf].v = __builtin_amdgcn_wmma_f32_16x16x32_bf16(
            false, af[mf].v, false, bfr[nf].v, (short)0, acc[mf][nf].v,
            false, false);

    wait_async0();          // next buffer fully landed
    __syncthreads();        // everyone done reading cur before it is refilled
  }

  for (int mf = 0; mf < 2; mf++) {
    for (int nf = 0; nf < 4; nf++) {
      for (int r = 0; r < 8; r++) {
        int row = bm*128 + wm*32 + mf*16 + r + 8*half;
        int col = bn*128 + wn*64 + nf*16 + lm;
        float v = acc[mf][nf].f[r];
        if (mode == 2) {
          outf[(size_t)row * CMD + col] = v + bias[col];
        } else {
          int b = row >> 11, s = row & (SEQ - 1);
          int h = col >> 6,  d = col & (DH - 1);
          size_t adr = (mode == 0)
            ? ((size_t)((b*HEADS + h)*SEQ + s) * DH + d)
            : ((size_t)((b*HEADS + h)*DH + d) * SEQ + s);
          outb[adr] = (__bf16)v;
        }
      }
    }
  }
}

// ---------------------------------------------------------------- attention
// Q,K in [b,h,s,d] bf16; V in [b,h,d,s] bf16; AO out [b,s,h*d] bf16.
// Block = 4 waves, one 64-query tile of one (b,h); flash softmax over S.
__global__ __launch_bounds__(128) void attn_bf16(
    const __bf16* __restrict__ Qb, const __bf16* __restrict__ Kb,
    const __bf16* __restrict__ Vtb, __bf16* __restrict__ AO)
{
  __shared__ alignas(16) __bf16 Ks[64][72];
  __shared__ alignas(16) __bf16 Vs[64][72];
  __shared__ alignas(16) __bf16 Ps[4][16][72];
  const int qt = blockIdx.x & 31;      // 32 query tiles of 64
  const int bh = blockIdx.x >> 5;      // 0..63
  const int t = threadIdx.x, lane = t & 31, wave = t >> 5;
  const int half = lane >> 4, lm = lane & 15;
  const __bf16* Qp = Qb  + (size_t)bh * SEQ * DH;
  const __bf16* Kp = Kb  + (size_t)bh * SEQ * DH;
  const __bf16* Vp = Vtb + (size_t)bh * DH * SEQ;

  // Q fragments (16 rows x 64 d) held in registers for whole kernel
  FragA aq[2];
  {
    const __bf16* qptr = Qp + (size_t)(qt*64 + wave*16 + lm) * DH + half*8;
    aq[0].q[0] = *(const vf4*)(qptr);
    aq[0].q[1] = *(const vf4*)(qptr + 16);
    aq[1].q[0] = *(const vf4*)(qptr + 32);
    aq[1].q[1] = *(const vf4*)(qptr + 48);
  }
  FragC accO[4];
  for (int i = 0; i < 4; i++)
    for (int r = 0; r < 8; r++) accO[i].f[r] = 0.f;
  float mrow[8], lrow[8];
  for (int r = 0; r < 8; r++) { mrow[r] = -3.0e38f; lrow[r] = 0.f; }

  const int lr  = t >> 1;
  const int seg = (t & 1) * 32;
  const unsigned ldsK = lds_off(&Ks[lr][seg]);
  const unsigned ldsV = lds_off(&Vs[lr][seg]);
  for (int kb = 0; kb < SEQ; kb += 64) {
    const __bf16* kg = Kp + (size_t)(kb + lr) * DH + seg;
    const __bf16* vg = Vp + (size_t)lr * SEQ + kb + seg;
    for (int c = 0; c < 4; c++) {
      async_b128(ldsK + c*16, kg + c*8);
      async_b128(ldsV + c*16, vg + c*8);
    }
    wait_async0();
    __syncthreads();

    // scores S = Q K^T  (16 q x 64 keys), contraction over d (2 wmma k-steps)
    FragC sc[4];
    for (int i = 0; i < 4; i++)
      for (int r = 0; r < 8; r++) sc[i].f[r] = 0.f;
    for (int kc = 0; kc < 2; kc++) {
      for (int nt = 0; nt < 4; nt++) {
        FragA bk;
        const __bf16* p = &Ks[nt*16 + lm][kc*32 + half*8];
        bk.q[0] = *(const vf4*)p;
        bk.q[1] = *(const vf4*)(p + 16);
        sc[nt].v = __builtin_amdgcn_wmma_f32_16x16x32_bf16(
            false, aq[kc].v, false, bk.v, (short)0, sc[nt].v, false, false);
      }
    }

    // online softmax: row stats replicated across the 16 lanes of each half
    for (int nt = 0; nt < 4; nt++)
      for (int r = 0; r < 8; r++) sc[nt].f[r] *= 0.125f;   // SCALE = D^-0.5
    for (int r = 0; r < 8; r++) {
      float mx = fmaxf(fmaxf(sc[0].f[r], sc[1].f[r]),
                       fmaxf(sc[2].f[r], sc[3].f[r]));
      mx = fmaxf(mx, __shfl_xor(mx, 1, 32));
      mx = fmaxf(mx, __shfl_xor(mx, 2, 32));
      mx = fmaxf(mx, __shfl_xor(mx, 4, 32));
      mx = fmaxf(mx, __shfl_xor(mx, 8, 32));
      float nm = fmaxf(mrow[r], mx);
      float corr = __expf(mrow[r] - nm);
      float rs = 0.f;
      for (int nt = 0; nt < 4; nt++) {
        float p = __expf(sc[nt].f[r] - nm);
        sc[nt].f[r] = p; rs += p;
      }
      rs += __shfl_xor(rs, 1, 32);
      rs += __shfl_xor(rs, 2, 32);
      rs += __shfl_xor(rs, 4, 32);
      rs += __shfl_xor(rs, 8, 32);
      lrow[r] = lrow[r] * corr + rs;
      mrow[r] = nm;
      for (int dt = 0; dt < 4; dt++) accO[dt].f[r] *= corr;
    }

    // restage P (C-layout) through LDS to get A-fragment layout
    for (int nt = 0; nt < 4; nt++)
      for (int r = 0; r < 8; r++)
        Ps[wave][r + 8*half][nt*16 + lm] = (__bf16)sc[nt].f[r];
    asm volatile("s_wait_dscnt 0" ::: "memory");

    // O += P V  (contraction over 64 keys = 2 wmma k-steps)
    for (int kc = 0; kc < 2; kc++) {
      FragA ap;
      const __bf16* pp = &Ps[wave][lm][kc*32 + half*8];
      ap.q[0] = *(const vf4*)pp;
      ap.q[1] = *(const vf4*)(pp + 16);
      for (int dt = 0; dt < 4; dt++) {
        FragA bv;
        const __bf16* vv = &Vs[dt*16 + lm][kc*32 + half*8];
        bv.q[0] = *(const vf4*)vv;
        bv.q[1] = *(const vf4*)(vv + 16);
        accO[dt].v = __builtin_amdgcn_wmma_f32_16x16x32_bf16(
            false, ap.v, false, bv.v, (short)0, accO[dt].v, false, false);
      }
    }
    __syncthreads();
  }

  const int b = bh >> 4, h = bh & 15;
  for (int r = 0; r < 8; r++) {
    float inv = 1.0f / lrow[r];
    int row = qt*64 + wave*16 + r + 8*half;
    for (int dt = 0; dt < 4; dt++) {
      int col = h*DH + dt*16 + lm;
      AO[(size_t)(b*SEQ + row) * CMD + col] = (__bf16)(accO[dt].f[r] * inv);
    }
  }
}

// ---------------------------------------------------------------- launch
extern "C" void kernel_launch(void* const* d_in, const int* in_sizes, int n_in,
                              void* d_out, int out_size, void* d_ws, size_t ws_size,
                              hipStream_t stream) {
  const float* hs = (const float*)d_in[0];
  const float* Wq = (const float*)d_in[1];
  const float* Wk = (const float*)d_in[2];
  const float* Wv = (const float*)d_in[3];
  const float* Wo = (const float*)d_in[4];
  const float* bo = (const float*)d_in[5];
  float* out = (float*)d_out;
  char* ws = (char*)d_ws;
  const size_t MB = 1ull << 20;
  __bf16* Xb  = (__bf16*)(ws + 0);        // 16 MB  x in bf16
  __bf16* WqT = (__bf16*)(ws + 16*MB);    //  2 MB  each, transposed bf16
  __bf16* WkT = (__bf16*)(ws + 18*MB);
  __bf16* WvT = (__bf16*)(ws + 20*MB);
  __bf16* WoT = (__bf16*)(ws + 22*MB);
  __bf16* Qb  = (__bf16*)(ws + 24*MB);    // 16 MB  [b,h,s,d]
  __bf16* Kb  = (__bf16*)(ws + 40*MB);    // 16 MB  [b,h,s,d]
  __bf16* Vtb = (__bf16*)(ws + 56*MB);    // 16 MB  [b,h,d,s]
  __bf16* AO  = (__bf16*)(ws + 72*MB);    // 16 MB  [b,s,h*d]

  cvt_f32_bf16<<<(NTOK * CMD) / 2048, 256, 0, stream>>>(hs, Xb);
  transpose_cvt<<<1024, 256, 0, stream>>>(Wq, WqT);
  transpose_cvt<<<1024, 256, 0, stream>>>(Wk, WkT);
  transpose_cvt<<<1024, 256, 0, stream>>>(Wv, WvT);
  transpose_cvt<<<1024, 256, 0, stream>>>(Wo, WoT);

  gemm_bf16<<<512, 256, 0, stream>>>(Xb, WqT, Qb,  nullptr, nullptr, 0);
  gemm_bf16<<<512, 256, 0, stream>>>(Xb, WkT, Kb,  nullptr, nullptr, 0);
  gemm_bf16<<<512, 256, 0, stream>>>(Xb, WvT, Vtb, nullptr, nullptr, 1);

  attn_bf16<<<BATCH * HEADS * (SEQ / 64), 128, 0, stream>>>(Qb, Kb, Vtb, AO);

  gemm_bf16<<<512, 256, 0, stream>>>(AO, WoT, nullptr, out, bo, 2);
}
